// ScannedRNN_29686813950314
// MI455X (gfx1250) — matrix operand order repaired and checked by reading, not compiled, not run on hardware
//
#include <hip/hip_runtime.h>
#include <hip/hip_bf16.h>

// ---------------------------------------------------------------------------
// GRU scan (ScannedRNN) for MI455X / gfx1250.
//
// Roofline reasoning:
//   - ~103 GFLOP total; half (x@Wi) is parallel over T, half (h@Wh) sequential.
//   - Per-step state (h, gates, weights) < 2 MB -> lives in 192 MB L2.
//   - HBM traffic = read ins (128 MB) + write ys (128 MB) ~ 11 us @ 23.3 TB/s.
//   => latency-bound sequential GEMMs. Use ONE persistent kernel, bf16 WMMA
//      (v_wmma_f32_16x16x32_bf16, f32 accumulate) for the matrix math, and
//      device-wide barriers between the GEMM and elementwise phases of each
//      of the 1024 steps. Everything stays hot in L2.
//
// Weight packing WT[1024][512] (bf16, N-major so B-fragments are contiguous):
//   n in [  0,512): k<256 -> Wi[k][n],      k>=256 -> Wh[k-256][n]   (r,z: K=512)
//   n in [512,768): k<256 -> Wi[k][n],      k>=256 -> 0              (i_n: K=256, A=x)
//   n in [768,1024): k<256 -> Wh[k][n-256], k>=256 -> 0              (h_n: K=256, A=h)
// A matrix per step: abuf[128][512] bf16 = [ x_t | masked h ].
// G[128][1024] f32 gate buffer: cols = r_pre | z_pre | i_n | h_n.
// ---------------------------------------------------------------------------

#define T_STEPS 1024
#define BATCH   128
#define HID     256
#define NCOL    1024
#define KDIM    512
#define NBLK    64
#define NTHR    256
#define GTHREADS (NBLK * NTHR)   // 16384

typedef __attribute__((ext_vector_type(16))) __bf16      v16bf;
typedef __attribute__((ext_vector_type(8)))  float        v8f;
typedef __attribute__((ext_vector_type(4)))  unsigned int u32x4;

union Frag { v16bf v; u32x4 q[2]; };

__device__ __forceinline__ void grid_barrier(unsigned* cnt, unsigned* gen, unsigned nb) {
    __threadfence();                 // make this thread's global stores device-visible
    __syncthreads();
    if (threadIdx.x == 0) {
        unsigned g = __hip_atomic_load(gen, __ATOMIC_RELAXED, __HIP_MEMORY_SCOPE_AGENT);
        unsigned old = __hip_atomic_fetch_add(cnt, 1u, __ATOMIC_ACQ_REL, __HIP_MEMORY_SCOPE_AGENT);
        if (old == nb - 1u) {
            __hip_atomic_store(cnt, 0u, __ATOMIC_RELAXED, __HIP_MEMORY_SCOPE_AGENT);
            __hip_atomic_store(gen, g + 1u, __ATOMIC_RELEASE, __HIP_MEMORY_SCOPE_AGENT);
        } else {
            while (__hip_atomic_load(gen, __ATOMIC_ACQUIRE, __HIP_MEMORY_SCOPE_AGENT) == g)
                __builtin_amdgcn_s_sleep(2);
        }
    }
    __syncthreads();
}

__device__ __forceinline__ float sigmoidf_fast(float x) {
    return 1.0f / (1.0f + __expf(-x));
}

// One-time prep: pack WT (bf16), initial A buffer, initial h buffer.
__global__ void __launch_bounds__(NTHR)
gru_prep_kernel(const float* __restrict__ Wi, const float* __restrict__ Wh,
                const float* __restrict__ ins, const unsigned char* __restrict__ resets,
                const float* __restrict__ h0,
                __bf16* __restrict__ WT, __bf16* __restrict__ abuf,
                float* __restrict__ hbuf) {
    int idx = blockIdx.x * NTHR + threadIdx.x;
    const int WT_ELEMS = NCOL * KDIM;                 // 524288
    if (idx < WT_ELEMS) {
        int n = idx >> 9;          // 0..1023
        int k = idx & 511;         // 0..511
        float v;
        if (n < 512)       v = (k < 256) ? Wi[k * 768 + n] : Wh[(k - 256) * 768 + n];
        else if (n < 768)  v = (k < 256) ? Wi[k * 768 + n] : 0.0f;
        else               v = (k < 256) ? Wh[k * 768 + (n - 256)] : 0.0f;
        WT[n * KDIM + k] = (__bf16)v;
    } else if (idx < WT_ELEMS + BATCH * HID) {
        int e = idx - WT_ELEMS;
        int b = e >> 8;            // 0..127
        int c = e & 255;           // 0..255
        float x0 = ins[b * HID + c];                  // t = 0
        float h  = h0[b * HID + c];
        float hm = resets[b] ? 0.0f : h;
        abuf[b * KDIM + c]       = (__bf16)x0;
        abuf[b * KDIM + HID + c] = (__bf16)hm;
        hbuf[b * HID + c] = h;
    }
}

// Persistent scan kernel: 64 blocks x 256 threads (512 wave32 waves).
__global__ void __launch_bounds__(NTHR)
gru_scan_kernel(const float* __restrict__ ins, const unsigned char* __restrict__ resets,
                const float* __restrict__ bi, const float* __restrict__ bhn,
                const __bf16* __restrict__ WT, __bf16* __restrict__ abuf,
                float* __restrict__ hbuf, float* __restrict__ G,
                unsigned* __restrict__ bar, float* __restrict__ out) {
    const int lane = threadIdx.x & 31;
    const int wave = threadIdx.x >> 5;
    const int w    = blockIdx.x * 8 + wave;       // 0..511 : one 16x16 output tile
    const int gid  = blockIdx.x * NTHR + threadIdx.x;

    // Tile coordinates in the [128 x 1024] gate matrix.
    const int nt = w >> 3;                        // 0..63
    const int mt = w & 7;                         // 0..7
    const int m0 = mt << 4;
    const int n0 = nt << 4;
    const int kChunks = (nt < 32) ? 16 : 8;       // r,z: K=512 ; i_n/h_n: K=256
    const int aCol0   = (nt >= 48) ? HID : 0;     // h_n tiles read the h half of A

    // Per-lane fragment base pointers (ISA wave32 WMMA layouts).
    const int arow = m0 + (lane & 15);
    const int kb   = (lane >> 4) << 3;            // A: 0 or 8
    const __bf16* aPtr = abuf + arow * KDIM + aCol0 + kb;
    const int bcol = n0 + (lane & 15);
    const __bf16* bPtr = WT + bcol * KDIM + ((lane >> 4) << 3);   // B: +0 or +8
    float* gOut = G + (m0 + ((lane >> 4) << 3)) * NCOL + n0 + (lane & 15);

    unsigned* cnt = bar;
    unsigned* gen = bar + 1;

    for (int t = 0; t < T_STEPS; ++t) {
        // ---------------- GEMM phase: G = [x|h] @ WT ----------------
        v8f acc = {};
        #pragma unroll 4
        for (int c = 0; c < kChunks; ++c) {
            const int kc = c << 5;                // K chunk of 32
            Frag a, b;
            a.q[0] = *(const u32x4*)(aPtr + kc);          // K = kc+kb   .. +8
            a.q[1] = *(const u32x4*)(aPtr + kc + 16);     // K = kc+16+kb.. +8
            b.q[0] = *(const u32x4*)(bPtr + kc);          // K = kc+lg*8 .. +8
            b.q[1] = *(const u32x4*)(bPtr + kc + 16);     // K = kc+16+lg*8.. +8
            acc = __builtin_amdgcn_wmma_f32_16x16x32_bf16(
                false, a.v, false, b.v, (short)0, acc, false, false);
        }
        #pragma unroll
        for (int r = 0; r < 8; ++r)
            gOut[r * NCOL] = acc[r];

        // Warm L2 with next step's x while the matrix pipe drains.
        if (t + 1 < T_STEPS)
            __builtin_prefetch(ins + (size_t)(t + 1) * (BATCH * HID) + gid * 2, 0, 1);

        grid_barrier(cnt, gen, NBLK);

        // ------------- Elementwise gates + next-step A prep -------------
        #pragma unroll
        for (int e = gid; e < BATCH * HID; e += GTHREADS) {
            const int b = e >> 8;
            const int c = e & 255;
            const float* gRow = G + b * NCOL;
            float rp  = gRow[c]             + bi[c];
            float zp  = gRow[HID + c]       + bi[HID + c];
            float inn = gRow[2 * HID + c]   + bi[2 * HID + c];
            float hn  = gRow[3 * HID + c];
            float r = sigmoidf_fast(rp);
            float z = sigmoidf_fast(zp);
            float n = tanhf(inn + r * (hn + bhn[c]));
            float hprev = hbuf[e];
            float hm = resets[t * BATCH + b] ? 0.0f : hprev;   // same mask GEMM used
            float hnew = (1.0f - z) * n + z * hm;

            out[(size_t)(BATCH * HID) + (size_t)t * (BATCH * HID) + e] = hnew;  // ys[t]
            hbuf[e] = hnew;

            if (t + 1 < T_STEPS) {
                float hmn = resets[(t + 1) * BATCH + b] ? 0.0f : hnew;
                abuf[b * KDIM + c]       = (__bf16)ins[(size_t)(t + 1) * (BATCH * HID) + e];
                abuf[b * KDIM + HID + c] = (__bf16)hmn;
            } else {
                out[e] = hnew;                                 // final carry h
            }
        }

        grid_barrier(cnt, gen, NBLK);
    }
}

extern "C" void kernel_launch(void* const* d_in, const int* in_sizes, int n_in,
                              void* d_out, int out_size, void* d_ws, size_t ws_size,
                              hipStream_t stream) {
    const float*         ins    = (const float*)d_in[0];
    const unsigned char* resets = (const unsigned char*)d_in[1];   // jnp.bool_ (1 byte)
    const float*         h0     = (const float*)d_in[2];
    const float*         Wi     = (const float*)d_in[3];
    const float*         Wh     = (const float*)d_in[4];
    const float*         bi     = (const float*)d_in[5];
    const float*         bhn    = (const float*)d_in[6];
    float* out = (float*)d_out;

    // Workspace layout (all offsets 256B-aligned):
    char* ws = (char*)d_ws;
    unsigned* bar  = (unsigned*)ws;                                 //   256 B (barrier state)
    __bf16*   WT   = (__bf16*)(ws + 256);                           //  1024*512*2  = 1 MB
    __bf16*   abuf = (__bf16*)(ws + 256 + 1048576);                 //   128*512*2  = 128 KB
    float*    hbuf = (float*) (ws + 256 + 1048576 + 131072);        //   128*256*4  = 128 KB
    float*    G    = (float*) (ws + 256 + 1048576 + 131072 + 131072); // 128*1024*4 = 512 KB

    hipMemsetAsync(bar, 0, 256, stream);

    const int prepWork = NCOL * KDIM + BATCH * HID;                 // 557056
    gru_prep_kernel<<<(prepWork + NTHR - 1) / NTHR, NTHR, 0, stream>>>(
        Wi, Wh, ins, resets, h0, WT, abuf, hbuf);

    gru_scan_kernel<<<NBLK, NTHR, 0, stream>>>(
        ins, resets, bi, bhn, WT, abuf, hbuf, G, bar, out);
}
